// GQAttention_83159156785870
// MI455X (gfx1250) — matrix operand order repaired and checked by reading, not compiled, not run on hardware
//
#include <hip/hip_runtime.h>
#include <hip/hip_bf16.h>

typedef __attribute__((ext_vector_type(16))) _Float16 v16h;
typedef __attribute__((ext_vector_type(8)))  float    v8f;

#define HIDDEN 2048
#define NHEADS 16
#define NKV 4
#define HEAD_DIM 128
#define BB 2
#define TT 2048
#define MROWS (BB*TT)          // 4096
#define NEGINF (-1.0e30f)

// ---------------- WMMA helpers (gfx1250 wave32) ----------------

__device__ __forceinline__ v8f wmma16x16x32(v16h a, v16h b, v8f c) {
  return __builtin_amdgcn_wmma_f32_16x16x32_f16(
      /*neg_a=*/false, a, /*neg_b=*/false, b,
      /*c_mod=*/(short)0, c, /*reuse_a=*/false, /*reuse_b=*/false);
}

// A fragment: 16x32 f16, row-major source with leading dim ld (halves).
// lane m = lane&15 ; lanes>=16 take K-offsets +8 ; halves 8..15 take K+16.
__device__ __forceinline__ v16h load_a_frag(const _Float16* base, int ld) {
  const int lane = threadIdx.x & 31;
  const int m  = lane & 15;
  const int kb = (lane & 16) >> 1;            // 0 or 8
  const _Float16* p = base + (size_t)m * ld + kb;
  union { v16h v; uint4 u[2]; } r;
  r.u[0] = *(const uint4*)(p);                // K = kb + 0..7
  r.u[1] = *(const uint4*)(p + 16);           // K = 16 + kb + 0..7
  return r.v;
}

// B fragment: 32x16 f16, row-major (contraction-major) source, leading dim ld.
// lane n = lane&15 ; lanes>=16 take K-offsets +8 ; halves 8..15 take K+16.
__device__ __forceinline__ v16h load_b_frag(const _Float16* base, int ld) {
  const int lane = threadIdx.x & 31;
  const int n  = lane & 15;
  const int kb = (lane & 16) >> 1;
  v16h r;
#pragma unroll
  for (int s = 0; s < 8; ++s) {
    r[s]     = base[(size_t)(kb + s) * ld + n];
    r[8 + s] = base[(size_t)(16 + kb + s) * ld + n];
  }
  return r;
}

// ---------------- CDNA5 async global->LDS copy (ASYNCcnt path) ----------------
// LDS aperture occupies addr[63:32]; low 32 bits of a generic shared pointer
// are the LDS byte offset, which is what the instruction's VDST expects.

__device__ __forceinline__ uint32_t lds_lo(const void* p) {
  return (uint32_t)(uintptr_t)p;
}

__device__ __forceinline__ void async_copy_b128(uint32_t lds_off,
                                                const _Float16* gptr) {
  asm volatile("global_load_async_to_lds_b128 %0, %1, off"
               :: "v"(lds_off), "v"(gptr)
               : "memory");
}

__device__ __forceinline__ void wait_async0() {
  asm volatile("s_wait_asynccnt 0" ::: "memory");
}

// ---------------- fp32 -> f16 convert ----------------

typedef __attribute__((ext_vector_type(4))) _Float16 v4h;

__global__ void cvt_f32_f16_k(const float* __restrict__ in,
                              _Float16* __restrict__ out, int n4) {
  int i = blockIdx.x * blockDim.x + threadIdx.x;
  if (i < n4) {
    float4 f = ((const float4*)in)[i];
    v4h h;
    h[0] = (_Float16)f.x; h[1] = (_Float16)f.y;
    h[2] = (_Float16)f.z; h[3] = (_Float16)f.w;
    ((v4h*)out)[i] = h;
  }
}

// ---------------- tiled WMMA GEMM: C[M][N] = A[M][K] * B[K][N] ----------------
// 256 threads = 8 waves. Workgroup tile 128x128, wave tile 32x64.
// Double-buffered LDS filled by async global->LDS DMA (no VGPR staging).

template <bool OUTF32>
__global__ __launch_bounds__(256) void gemm_f16_k(
    const _Float16* __restrict__ A, const _Float16* __restrict__ B,
    void* __restrict__ C, int M, int N, int K) {
  __shared__ _Float16 As[2][128 * 32];
  __shared__ _Float16 Bs[2][32 * 128];
  const int tid  = threadIdx.x;
  const int lane = tid & 31;
  const int w    = tid >> 5;
  const int wm   = w >> 1;          // 0..3  -> m offset wm*32
  const int wn   = w & 1;           // 0..1  -> n offset wn*64
  const int m0   = blockIdx.y * 128;
  const int n0   = blockIdx.x * 128;

  v8f acc[2][4] = {};

  const int ar = tid >> 1, ac = (tid & 1) * 16;   // A tile 128x32: 16 halves/thread
  const int br = tid >> 3, bc = (tid & 7) * 16;   // B tile 32x128: 16 halves/thread

  // issue async fills for stage `kk` into buffer `buf`
  auto stage = [&](int buf, int kk) {
    _Float16* Ab = &As[buf][0];
    _Float16* Bb = &Bs[buf][0];
    const _Float16* ga = A + (size_t)(m0 + ar) * K + kk + ac;
    const _Float16* gb = B + (size_t)(kk + br) * N + n0 + bc;
    async_copy_b128(lds_lo(Ab + ar * 32 + ac),      ga);
    async_copy_b128(lds_lo(Ab + ar * 32 + ac + 8),  ga + 8);
    async_copy_b128(lds_lo(Bb + br * 128 + bc),     gb);
    async_copy_b128(lds_lo(Bb + br * 128 + bc + 8), gb + 8);
  };

  stage(0, 0);

  for (int kk = 0; kk < K; kk += 32) {
    const int cur = (kk >> 5) & 1;
    wait_async0();        // this wave's fills for buffer `cur` complete
    __syncthreads();      // all waves' fills complete
    if (kk + 32 < K) stage(cur ^ 1, kk + 32);

    v16h af[2], bf[4];
    af[0] = load_a_frag(&As[cur][(wm * 32 +  0) * 32], 32);
    af[1] = load_a_frag(&As[cur][(wm * 32 + 16) * 32], 32);
#pragma unroll
    for (int nt = 0; nt < 4; ++nt)
      bf[nt] = load_b_frag(&Bs[cur][wn * 64 + nt * 16], 128);
#pragma unroll
    for (int mt = 0; mt < 2; ++mt)
#pragma unroll
      for (int nt = 0; nt < 4; ++nt)
        acc[mt][nt] = wmma16x16x32(af[mt], bf[nt], acc[mt][nt]);
    // no trailing barrier: fragment DS reads are consumed before each wave's
    // next-iteration barrier, and a buffer is only refilled after that barrier.
  }

  // C layout: row = j + 8*(lane>=16), col = lane&15
  const int nn    = lane & 15;
  const int mhalf = (lane >> 4) << 3;
#pragma unroll
  for (int mt = 0; mt < 2; ++mt)
#pragma unroll
    for (int nt = 0; nt < 4; ++nt)
#pragma unroll
      for (int j = 0; j < 8; ++j) {
        int row = m0 + wm * 32 + mt * 16 + j + mhalf;
        int col = n0 + wn * 64 + nt * 16 + nn;
        if (OUTF32)
          ((float*)C)[(size_t)row * N + col] = acc[mt][nt][j];
        else
          ((_Float16*)C)[(size_t)row * N + col] = (_Float16)acc[mt][nt][j];
      }
}

// ---------------- RoPE ----------------

// In-place RoPE on q_h [B*T][NHEADS*128]
__global__ void rope_q_k(_Float16* __restrict__ q,
                         const float* __restrict__ cosb,
                         const float* __restrict__ sinb) {
  int idx = blockIdx.x * 256 + threadIdx.x;     // B*T*NHEADS*64 threads
  int d  = idx & 63;
  int h  = (idx >> 6) & (NHEADS - 1);
  int bt = idx >> 10;
  int t  = bt & (TT - 1);
  _Float16* p = q + (size_t)bt * (NHEADS * HEAD_DIM) + h * HEAD_DIM + d;
  float x1 = (float)p[0], x2 = (float)p[64];
  float c = cosb[t * 64 + d], s = sinb[t * 64 + d];
  p[0]  = (_Float16)(x1 * c - x2 * s);
  p[64] = (_Float16)(x2 * c + x1 * s);
}

// RoPE on K and write transposed kT[b][kv][d][t]
__global__ void rope_kT_k(const _Float16* __restrict__ k,
                          _Float16* __restrict__ kT,
                          const float* __restrict__ cosb,
                          const float* __restrict__ sinb) {
  int t   = blockIdx.x * 256 + threadIdx.x;
  int lin = blockIdx.y;                          // B*NKV*64
  int d  = lin & 63;
  int kv = (lin >> 6) & (NKV - 1);
  int b  = lin >> 8;
  const _Float16* p = k + (size_t)(b * TT + t) * (NKV * HEAD_DIM) + kv * HEAD_DIM + d;
  float x1 = (float)p[0], x2 = (float)p[64];
  float c = cosb[t * 64 + d], s = sinb[t * 64 + d];
  _Float16* o = kT + (size_t)((b * NKV + kv) * HEAD_DIM + d) * TT + t;
  o[0]               = (_Float16)(x1 * c - x2 * s);
  o[64 * (size_t)TT] = (_Float16)(x2 * c + x1 * s);
}

// ---------------- flash attention ----------------
// grid (T/128, NHEADS, B), 256 threads (8 waves). Wave owns 16 query rows.

__global__ __launch_bounds__(256) void flash_attn_k(
    const _Float16* __restrict__ q,   // [B*T][2048] roped
    const _Float16* __restrict__ kT,  // [B][NKV][128][T] roped
    const _Float16* __restrict__ v,   // [B*T][512]
    _Float16* __restrict__ out) {     // [B*T][2048]
  const int lane  = threadIdx.x & 31;
  const int w     = threadIdx.x >> 5;
  const int nn    = lane & 15;
  const int mhalf = (lane >> 4) << 3;
  const int h  = blockIdx.y;
  const int b  = blockIdx.z;
  const int kv = h >> 2;                         // NHEADS/NKV = 4
  const int qbase = blockIdx.x * 128 + w * 16;
  const float sc = 0.08838834764831845f;         // 1/sqrt(128)

  const _Float16* qp  = q  + ((size_t)(b * TT) + qbase) * (NHEADS * HEAD_DIM) + h * HEAD_DIM;
  const _Float16* kTp = kT + (size_t)((b * NKV + kv) * HEAD_DIM) * TT;
  const _Float16* vp  = v  + (size_t)(b * TT) * (NKV * HEAD_DIM) + kv * HEAD_DIM;

  __shared__ _Float16 Pls[8][16 * 32];           // per-wave P staging (1KB each)
  _Float16* pw = &Pls[w][0];

  v16h aq[4];
#pragma unroll
  for (int dc = 0; dc < 4; ++dc)
    aq[dc] = load_a_frag(qp + dc * 32, NHEADS * HEAD_DIM);

  v8f acc[8] = {};
  float mrow[8], lsum[8];
#pragma unroll
  for (int j = 0; j < 8; ++j) { mrow[j] = NEGINF; lsum[j] = 0.0f; }

  for (int kb = 0; kb < qbase + 16; kb += 32) {
    v8f s0 = {}, s1 = {};
#pragma unroll
    for (int dc = 0; dc < 4; ++dc) {
      v16h b0 = load_b_frag(kTp + (size_t)(dc * 32) * TT + kb,      TT);
      v16h b1 = load_b_frag(kTp + (size_t)(dc * 32) * TT + kb + 16, TT);
      s0 = wmma16x16x32(aq[dc], b0, s0);
      s1 = wmma16x16x32(aq[dc], b1, s1);
    }
#pragma unroll
    for (int j = 0; j < 8; ++j) {
      const int qi = qbase + j + mhalf;
      float v0 = s0[j] * sc; if (kb + nn      > qi) v0 = NEGINF;
      float v1 = s1[j] * sc; if (kb + 16 + nn > qi) v1 = NEGINF;
      // row max over 32 keys (reduce within 16-lane half)
      float mx = fmaxf(v0, v1);
      mx = fmaxf(mx, __shfl_xor(mx, 1));
      mx = fmaxf(mx, __shfl_xor(mx, 2));
      mx = fmaxf(mx, __shfl_xor(mx, 4));
      mx = fmaxf(mx, __shfl_xor(mx, 8));
      float nm = fmaxf(mrow[j], mx);
      float fj = __expf(mrow[j] - nm);
      mrow[j] = nm;
      float p0 = __expf(v0 - nm);
      float p1 = __expf(v1 - nm);
      float ps = p0 + p1;
      ps += __shfl_xor(ps, 1);
      ps += __shfl_xor(ps, 2);
      ps += __shfl_xor(ps, 4);
      ps += __shfl_xor(ps, 8);
      lsum[j] = lsum[j] * fj + ps;
#pragma unroll
      for (int nt = 0; nt < 8; ++nt) acc[nt][j] *= fj;
      // stage P (16x32 row-major f16) for A-fragment reload
      pw[(j + mhalf) * 32 + nn]      = (_Float16)p0;
      pw[(j + mhalf) * 32 + 16 + nn] = (_Float16)p1;
    }
    asm volatile("s_wait_dscnt 0" ::: "memory");
    v16h ap = load_a_frag(pw, 32);
#pragma unroll
    for (int nt = 0; nt < 8; ++nt) {
      v16h bv = load_b_frag(vp + (size_t)kb * (NKV * HEAD_DIM) + nt * 16,
                            NKV * HEAD_DIM);
      acc[nt] = wmma16x16x32(ap, bv, acc[nt]);
    }
  }

  _Float16* op = out + ((size_t)(b * TT) + qbase) * (NHEADS * HEAD_DIM) + h * HEAD_DIM;
#pragma unroll
  for (int j = 0; j < 8; ++j) {
    float inv = 1.0f / lsum[j];
#pragma unroll
    for (int nt = 0; nt < 8; ++nt)
      op[(size_t)(j + mhalf) * (NHEADS * HEAD_DIM) + nt * 16 + nn] =
          (_Float16)(acc[nt][j] * inv);
  }
}

// ---------------- launch ----------------

extern "C" void kernel_launch(void* const* d_in, const int* in_sizes, int n_in,
                              void* d_out, int out_size, void* d_ws, size_t ws_size,
                              hipStream_t stream) {
  const float* x   = (const float*)d_in[0];
  const float* cosb= (const float*)d_in[1];
  const float* sinb= (const float*)d_in[2];
  const float* wq  = (const float*)d_in[3];
  const float* wk  = (const float*)d_in[4];
  const float* wv  = (const float*)d_in[5];
  const float* wo  = (const float*)d_in[6];

  _Float16* ws = (_Float16*)d_ws;
  // workspace layout (element offsets, halves)
  _Float16* x_h   = ws;                              // 4096*2048
  _Float16* wq_h  = x_h  + (size_t)MROWS * HIDDEN;   // 2048*2048
  _Float16* wk_h  = wq_h + (size_t)HIDDEN * HIDDEN;  // 2048*512
  _Float16* wv_h  = wk_h + (size_t)HIDDEN * 512;
  _Float16* wo_h  = wv_h + (size_t)HIDDEN * 512;     // 2048*2048
  _Float16* q_h   = wo_h + (size_t)HIDDEN * HIDDEN;  // 4096*2048
  _Float16* k_h   = q_h  + (size_t)MROWS * HIDDEN;   // 4096*512
  _Float16* v_h   = k_h  + (size_t)MROWS * 512;
  _Float16* kT_h  = v_h  + (size_t)MROWS * 512;      // 2*4*128*2048
  _Float16* at_h  = kT_h + (size_t)BB * NKV * HEAD_DIM * TT; // 4096*2048

  // 1) convert fp32 -> f16
  auto cvt = [&](const float* in, _Float16* o, size_t n) {
    int n4 = (int)(n / 4);
    cvt_f32_f16_k<<<(n4 + 255) / 256, 256, 0, stream>>>(in, o, n4);
  };
  cvt(x,  x_h,  (size_t)MROWS * HIDDEN);
  cvt(wq, wq_h, (size_t)HIDDEN * HIDDEN);
  cvt(wk, wk_h, (size_t)HIDDEN * 512);
  cvt(wv, wv_h, (size_t)HIDDEN * 512);
  cvt(wo, wo_h, (size_t)HIDDEN * HIDDEN);

  // 2) QKV projections
  gemm_f16_k<false><<<dim3(HIDDEN / 128, MROWS / 128), 256, 0, stream>>>(
      x_h, wq_h, (void*)q_h, MROWS, HIDDEN, HIDDEN);
  gemm_f16_k<false><<<dim3(512 / 128, MROWS / 128), 256, 0, stream>>>(
      x_h, wk_h, (void*)k_h, MROWS, 512, HIDDEN);
  gemm_f16_k<false><<<dim3(512 / 128, MROWS / 128), 256, 0, stream>>>(
      x_h, wv_h, (void*)v_h, MROWS, 512, HIDDEN);

  // 3) RoPE (q in place; k -> transposed kT)
  {
    int nq = MROWS * NHEADS * 64;                 // threads
    rope_q_k<<<nq / 256, 256, 0, stream>>>(q_h, cosb, sinb);
    rope_kT_k<<<dim3(TT / 256, BB * NKV * 64), 256, 0, stream>>>(
        k_h, kT_h, cosb, sinb);
  }

  // 4) flash attention
  flash_attn_k<<<dim3(TT / 128, NHEADS, BB), 256, 0, stream>>>(
      q_h, kT_h, v_h, at_h);

  // 5) output projection -> fp32 d_out
  gemm_f16_k<true><<<dim3(HIDDEN / 128, MROWS / 128), 256, 0, stream>>>(
      at_h, wo_h, d_out, MROWS, HIDDEN, HIDDEN);

  (void)in_sizes; (void)n_in; (void)out_size; (void)ws_size;
}
